// ClauseLevelMPN_39084202393955
// MI455X (gfx1250) — compile-verified, matrix operand order checked by reading, not run on hardware
//
#include <hip/hip_runtime.h>
#include <math.h>

// ---------------------------------------------------------------------------
// Problem constants (from the reference)
// ---------------------------------------------------------------------------
static constexpr int   Ncl   = 20000;   // clauses
static constexpr int   Hdim  = 256;     // hidden
static constexpr int   Llen  = 32;      // literals per clause
static constexpr int   FEAT  = 7;
static constexpr int   NEDGE = 640000;
static constexpr int   NLAY  = 2;

// ---------------------------------------------------------------------------
// WMMA types / helpers (gfx1250, wave32)
// ---------------------------------------------------------------------------
typedef __attribute__((ext_vector_type(16))) __bf16 bf16x16;
typedef __attribute__((ext_vector_type(8)))  float  v8f;

union FragU { bf16x16 v; uint4 q[2]; };

static __device__ inline v8f wmma_bf16(bf16x16 a, bf16x16 b, v8f c) {
    // v_wmma_f32_16x16x32_bf16  D = A*B + C
    return __builtin_amdgcn_wmma_f32_16x16x32_bf16(
        /*neg_a=*/false, a, /*neg_b=*/false, b,
        /*c_mod=*/(short)0, c, /*reuse_a=*/false, /*reuse_b=*/false);
}

static __device__ inline unsigned short f2bf(float f) {
    unsigned u = __float_as_uint(f);
    unsigned r = u + 0x7FFFu + ((u >> 16) & 1u);   // round-to-nearest-even
    return (unsigned short)(r >> 16);
}

// A fragment (16x32 bf16) from an LDS tile of bf16 rows (row-major, stride in elems)
// ISA layout: lane m=lane&15; K base = (lane<16?0:8); elems 0..7 = K base..base+7,
// elems 8..15 = K base+16..base+23  -> two contiguous 16B chunks 32B apart.
static __device__ inline bf16x16 load_a_frag(const unsigned short* lds,
                                             int rowBase, int stride, int s) {
    int lane  = threadIdx.x & 31;
    int m     = lane & 15;
    int base1 = (lane & 16) ? 8 : 0;
    const unsigned short* p = lds + (rowBase + m) * stride + s * 32 + base1;
    FragU f;
    f.q[0] = *(const uint4*)(p);
    f.q[1] = *(const uint4*)(p + 16);
    return f.v;
}

// B fragment (32x16 bf16) from packed weights: 32B contiguous per lane.
static __device__ inline bf16x16 load_b_frag(const unsigned short* pk,
                                             int nt, int ksteps, int s) {
    int lane = threadIdx.x & 31;
    const uint4* p = (const uint4*)(pk + (size_t)(((nt * ksteps) + s) * 32 + lane) * 16);
    FragU f;
    f.q[0] = p[0];
    f.q[1] = p[1];
    return f.v;
}

static __device__ inline float wred32(float v) {
    #pragma unroll
    for (int o = 16; o > 0; o >>= 1) v += __shfl_xor(v, o, 32);
    return v;
}

// Software-pipelined 4-tile WMMA pass over one LDS A tile and packed B weights.
// Prefetches the next K-step's A + 4 B fragments into fresh registers while the
// current step's 4 WMMAs issue (kills the load->wait->wmma serialization).
template <int KSTEPS>
static __device__ inline void wmma_pass(const unsigned short* aTile, int aStride,
                                        int rowBase, const unsigned short* Bpk,
                                        int g4, v8f acc[4]) {
    bf16x16 aCur = load_a_frag(aTile, rowBase, aStride, 0);
    bf16x16 b0 = load_b_frag(Bpk, g4 + 0, KSTEPS, 0);
    bf16x16 b1 = load_b_frag(Bpk, g4 + 1, KSTEPS, 0);
    bf16x16 b2 = load_b_frag(Bpk, g4 + 2, KSTEPS, 0);
    bf16x16 b3 = load_b_frag(Bpk, g4 + 3, KSTEPS, 0);
    #pragma unroll
    for (int s = 0; s < KSTEPS; ++s) {
        // branch-free prefetch (last iteration redundantly reloads step s)
        int sn = (s + 1 < KSTEPS) ? (s + 1) : s;
        bf16x16 aN  = load_a_frag(aTile, rowBase, aStride, sn);
        bf16x16 n0 = load_b_frag(Bpk, g4 + 0, KSTEPS, sn);
        bf16x16 n1 = load_b_frag(Bpk, g4 + 1, KSTEPS, sn);
        bf16x16 n2 = load_b_frag(Bpk, g4 + 2, KSTEPS, sn);
        bf16x16 n3 = load_b_frag(Bpk, g4 + 3, KSTEPS, sn);
        acc[0] = wmma_bf16(aCur, b0, acc[0]);
        acc[1] = wmma_bf16(aCur, b1, acc[1]);
        acc[2] = wmma_bf16(aCur, b2, acc[2]);
        acc[3] = wmma_bf16(aCur, b3, acc[3]);
        aCur = aN; b0 = n0; b1 = n1; b2 = n2; b3 = n3;
    }
}

// ---------------------------------------------------------------------------
// Weight packer: f32 row-major W[k, col] (row stride ld, column offset coloff)
//  -> bf16 fragment-contiguous layout: idx = (((nt*KSTEPS)+s)*32 + lane)*16 + e
// ---------------------------------------------------------------------------
__global__ __launch_bounds__(256) void pack_kernel(const float* __restrict__ src,
                                                   int ld, int coloff, int ksteps,
                                                   unsigned short* __restrict__ dst) {
    int idx  = blockIdx.x * 256 + threadIdx.x;
    int e    = idx & 15;
    int lane = (idx >> 4) & 31;
    int rest = idx >> 9;
    int s    = rest % ksteps;
    int nt   = rest / ksteps;
    int k    = s * 32 + ((lane & 16) ? 8 : 0) + ((e & 8) ? 16 : 0) + (e & 7);
    int col  = coloff + nt * 16 + (lane & 15);
    dst[idx] = f2bf(src[(size_t)k * ld + col]);
}

// ---------------------------------------------------------------------------
// Encoder stage 1: hidden = relu(cf @ pe_w1 + pe_b1)  (K=7, scalar; store bf16)
// ---------------------------------------------------------------------------
__global__ __launch_bounds__(256) void pe1_kernel(const float* __restrict__ cf,
                                                  const float* __restrict__ w1,
                                                  const float* __restrict__ b1,
                                                  unsigned short* __restrict__ hidden) {
    int n = blockIdx.x, h = threadIdx.x;
    float acc = b1[h];
    #pragma unroll
    for (int f = 0; f < FEAT; ++f) acc += cf[(size_t)n * FEAT + f] * w1[f * Hdim + h];
    hidden[(size_t)n * Hdim + h] = f2bf(fmaxf(acc, 0.0f));
}

// ---------------------------------------------------------------------------
// Generic WMMA GEMM:  C[nrows x 256] = act(A[nrows x K]_bf16 @ Wpk + bias)
// Block: 256 threads = 8 waves, covers 32 rows x 256 cols.
// ACT: 0=none 1=scale 2=sigmoid 3=relu
// ---------------------------------------------------------------------------
template <int KSTEPS, int ACT>
__global__ __launch_bounds__(256) void gemm_wmma(const unsigned short* __restrict__ A,
                                                 const unsigned short* __restrict__ Bpk,
                                                 const float* __restrict__ bias,
                                                 float* __restrict__ outF,
                                                 unsigned short* __restrict__ outB,
                                                 float scale) {
    constexpr int K   = KSTEPS * 32;
    constexpr int STR = K + 8;                // LDS pad to dodge bank conflicts
    __shared__ unsigned short aTile[32 * STR];

    int row0 = blockIdx.x * 32;

    // Stage A tile (bf16) into LDS, 16B per thread per iteration.
    constexpr int NLD = K / 64;               // uint4 loads per thread
    #pragma unroll
    for (int i = 0; i < NLD; ++i) {
        int lin = i * 256 + threadIdx.x;      // uint4 index
        int r   = lin / (K / 8);
        int c8  = lin % (K / 8);
        uint4 d = *(const uint4*)(A + (size_t)(row0 + r) * K + c8 * 8);
        *(uint4*)(aTile + r * STR + c8 * 8) = d;
    }
    __syncthreads();

    int wave = threadIdx.x >> 5, lane = threadIdx.x & 31;
    int mt = wave & 1;           // row tile (0..1)
    int g  = wave >> 1;          // column group (0..3), 4 n-tiles each

    v8f acc[4];
    #pragma unroll
    for (int j = 0; j < 4; ++j)
        #pragma unroll
        for (int r = 0; r < 8; ++r) acc[j][r] = 0.0f;

    wmma_pass<KSTEPS>(aTile, STR, mt * 16, Bpk, g * 4, acc);

    #pragma unroll
    for (int j = 0; j < 4; ++j) {
        int col = (g * 4 + j) * 16 + (lane & 15);
        float bv = bias[col];
        #pragma unroll
        for (int r = 0; r < 8; ++r) {
            float v = acc[j][r] + bv;
            if (ACT == 1) v *= scale;
            if (ACT == 2) v = 1.0f / (1.0f + __expf(-v));
            if (ACT == 3) v = fmaxf(v, 0.0f);
            int row = row0 + mt * 16 + ((lane >> 4) << 3) + r;
            if (outF) outF[(size_t)row * Hdim + col] = v;
            if (outB) outB[(size_t)row * Hdim + col] = f2bf(v);
        }
    }
}

// ---------------------------------------------------------------------------
// Fused per-clause attention: one workgroup per clause.
//   K = lit @ wk + bk ; V = lit @ wv + bv    (WMMA, bf16 in / f32 out)
//   scores = q . K per head, masked softmax over L=32, ctx = attn @ V
// Also computes lit_mean and the all-masked flag.
// Two sequential pipelined WMMA passes (K then V) keep VGPR pressure sane.
// ---------------------------------------------------------------------------
__global__ __launch_bounds__(256) void attn_kernel(const float* __restrict__ lit,
                                                   const unsigned char* __restrict__ mask,
                                                   const float* __restrict__ qv,
                                                   const unsigned short* __restrict__ pwk,
                                                   const unsigned short* __restrict__ pwv,
                                                   const float* __restrict__ attn_in_b,
                                                   unsigned short* __restrict__ ctx_bf,
                                                   float* __restrict__ litmean,
                                                   float* __restrict__ flags) {
    __shared__ unsigned short litA[32 * 264];
    __shared__ float kbuf[32 * 264];
    __shared__ float vbuf[32 * 264];
    __shared__ float attnw[4 * 32];

    int n = blockIdx.x, t = threadIdx.x;

    // Mask bits (all threads compute identical value; reads hit L2/L0)
    unsigned m32 = 0;
    #pragma unroll 4
    for (int r = 0; r < Llen; ++r)
        m32 |= (mask[(size_t)n * Llen + r] ? (1u << r) : 0u);
    unsigned allowed = m32 ? m32 : 1u;          // kpm[:,0] forced valid if all masked
    float inv = 1.0f / fmaxf((float)__popc(m32), 1.0f);

    // Load literal tile -> LDS bf16, fused masked column sum for lit_mean
    {
        float sum = 0.0f;
        const float* lp = lit + (size_t)n * Llen * Hdim + t;
        #pragma unroll 4
        for (int r = 0; r < Llen; ++r) {
            float f = lp[(size_t)r * Hdim];
            litA[r * 264 + t] = f2bf(f);
            if (m32 & (1u << r)) sum += f;
        }
        litmean[(size_t)n * Hdim + t] = sum * inv;
        if (t == 0) flags[n] = (m32 == 0u) ? 1.0f : 0.0f;
    }
    __syncthreads();

    int wave = t >> 5, lane = t & 31;
    int mt = wave & 1, g = wave >> 1;
    const float* bk = attn_in_b + Hdim;       // attn_in_b[H : 2H]
    const float* bv = attn_in_b + 2 * Hdim;   // attn_in_b[2H : 3H]

    // Pass 1: K projection
    {
        v8f acc[4];
        #pragma unroll
        for (int j = 0; j < 4; ++j)
            #pragma unroll
            for (int r = 0; r < 8; ++r) acc[j][r] = 0.0f;
        wmma_pass<8>(litA, 264, mt * 16, pwk, g * 4, acc);
        #pragma unroll
        for (int j = 0; j < 4; ++j) {
            int col = (g * 4 + j) * 16 + (lane & 15);
            float bkv = bk[col];
            #pragma unroll
            for (int r = 0; r < 8; ++r) {
                int row = mt * 16 + ((lane >> 4) << 3) + r;
                kbuf[row * 264 + col] = acc[j][r] + bkv;
            }
        }
    }
    // Pass 2: V projection
    {
        v8f acc[4];
        #pragma unroll
        for (int j = 0; j < 4; ++j)
            #pragma unroll
            for (int r = 0; r < 8; ++r) acc[j][r] = 0.0f;
        wmma_pass<8>(litA, 264, mt * 16, pwv, g * 4, acc);
        #pragma unroll
        for (int j = 0; j < 4; ++j) {
            int col = (g * 4 + j) * 16 + (lane & 15);
            float bvv = bv[col];
            #pragma unroll
            for (int r = 0; r < 8; ++r) {
                int row = mt * 16 + ((lane >> 4) << 3) + r;
                vbuf[row * 264 + col] = acc[j][r] + bvv;
            }
        }
    }
    __syncthreads();

    // scores[h, l] = sum_d q[h,d] * K[l, h*64+d], with key-padding mask
    if (t < 128) {
        int h = t >> 5, l = t & 31;
        const float* qp = qv + (size_t)n * Hdim + h * 64;
        float s = 0.0f;
        #pragma unroll 8
        for (int d = 0; d < 64; ++d) s += qp[d] * kbuf[l * 264 + h * 64 + d];
        attnw[h * 32 + l] = (allowed & (1u << l)) ? s : -__builtin_inff();
    }
    __syncthreads();

    // softmax per head (tiny: 4 heads x 32)
    if (t < 4) {
        int h = t;
        float mx = -__builtin_inff();
        for (int l = 0; l < 32; ++l) mx = fmaxf(mx, attnw[h * 32 + l]);
        float se = 0.0f;
        for (int l = 0; l < 32; ++l) {
            float e = __expf(attnw[h * 32 + l] - mx);
            attnw[h * 32 + l] = e;
            se += e;
        }
        float is = 1.0f / se;
        for (int l = 0; l < 32; ++l) attnw[h * 32 + l] *= is;
    }
    __syncthreads();

    // ctx[d] = sum_l attn[h(d), l] * V[l, d]   -> bf16 for the out-proj GEMM
    {
        int d = t, h = d >> 6;
        float c = 0.0f;
        #pragma unroll 8
        for (int l = 0; l < Llen; ++l) c += attnw[h * 32 + l] * vbuf[l * 264 + d];
        ctx_bf[(size_t)n * Hdim + d] = f2bf(c);
    }
}

// ---------------------------------------------------------------------------
// LayerNorm kernels: 1 wave per row, 8 rows per block, shfl reductions.
// ---------------------------------------------------------------------------
__global__ __launch_bounds__(256) void attn_ln_kernel(const float* __restrict__ ctxp,
                                                      const float* __restrict__ enc,
                                                      const float* __restrict__ litmean,
                                                      const float* __restrict__ flags,
                                                      const float* __restrict__ g,
                                                      const float* __restrict__ b,
                                                      float* __restrict__ attnrepr,
                                                      unsigned short* __restrict__ comb) {
    int wave = threadIdx.x >> 5, lane = threadIdx.x & 31;
    int n = blockIdx.x * 8 + wave;
    bool zero = flags[n] != 0.0f;
    float vals[8], encv[8], s = 0.0f;
    #pragma unroll
    for (int j = 0; j < 8; ++j) {
        int c = lane + 32 * j;
        float e  = enc[(size_t)n * Hdim + c];
        float cp = zero ? 0.0f : ctxp[(size_t)n * Hdim + c];
        float tv = cp + e;
        vals[j] = tv; encv[j] = e; s += tv;
    }
    float mu = wred32(s) * (1.0f / Hdim);
    float vs = 0.0f;
    #pragma unroll
    for (int j = 0; j < 8; ++j) { float d = vals[j] - mu; vs += d * d; }
    float rinv = rsqrtf(wred32(vs) * (1.0f / Hdim) + 1e-5f);
    #pragma unroll
    for (int j = 0; j < 8; ++j) {
        int c = lane + 32 * j;
        float y = (vals[j] - mu) * rinv * g[c] + b[c] + litmean[(size_t)n * Hdim + c];
        attnrepr[(size_t)n * Hdim + c] = y;
        comb[(size_t)n * 512 + c]       = f2bf(y);
        comb[(size_t)n * 512 + 256 + c] = f2bf(encv[j]);
    }
}

__global__ __launch_bounds__(256) void fuse_ln_kernel(const float* __restrict__ ar,
                                                      const float* __restrict__ gate,
                                                      const float* __restrict__ tr,
                                                      const float* __restrict__ g,
                                                      const float* __restrict__ b,
                                                      float* __restrict__ x) {
    int wave = threadIdx.x >> 5, lane = threadIdx.x & 31;
    int n = blockIdx.x * 8 + wave;
    float vals[8], s = 0.0f;
    #pragma unroll
    for (int j = 0; j < 8; ++j) {
        int c = lane + 32 * j;
        size_t i = (size_t)n * Hdim + c;
        float gv = gate[i];
        float tv = gv * ar[i] + (1.0f - gv) * tr[i];
        vals[j] = tv; s += tv;
    }
    float mu = wred32(s) * (1.0f / Hdim);
    float vs = 0.0f;
    #pragma unroll
    for (int j = 0; j < 8; ++j) { float d = vals[j] - mu; vs += d * d; }
    float rinv = rsqrtf(wred32(vs) * (1.0f / Hdim) + 1e-5f);
    #pragma unroll
    for (int j = 0; j < 8; ++j) {
        int c = lane + 32 * j;
        x[(size_t)n * Hdim + c] = (vals[j] - mu) * rinv * g[c] + b[c];
    }
}

__global__ __launch_bounds__(256) void mp_ln_kernel(const float* __restrict__ x,
                                                    const float* __restrict__ upd,
                                                    const float* __restrict__ g,
                                                    const float* __restrict__ b,
                                                    float* __restrict__ out) {
    int wave = threadIdx.x >> 5, lane = threadIdx.x & 31;
    int n = blockIdx.x * 8 + wave;
    float vals[8], s = 0.0f;
    #pragma unroll
    for (int j = 0; j < 8; ++j) {
        int c = lane + 32 * j;
        size_t i = (size_t)n * Hdim + c;
        float tv = x[i] + upd[i];
        vals[j] = tv; s += tv;
    }
    float mu = wred32(s) * (1.0f / Hdim);
    float vs = 0.0f;
    #pragma unroll
    for (int j = 0; j < 8; ++j) { float d = vals[j] - mu; vs += d * d; }
    float rinv = rsqrtf(wred32(vs) * (1.0f / Hdim) + 1e-5f);
    #pragma unroll
    for (int j = 0; j < 8; ++j) {
        int c = lane + 32 * j;
        out[(size_t)n * Hdim + c] = (vals[j] - mu) * rinv * g[c] + b[c];
    }
}

// ---------------------------------------------------------------------------
// Message passing helpers
// ---------------------------------------------------------------------------
__global__ __launch_bounds__(256) void zero_kernel(float* __restrict__ p, size_t count) {
    size_t i = (size_t)blockIdx.x * 256 + threadIdx.x;
    if (i < count) p[i] = 0.0f;
}

__global__ __launch_bounds__(256) void scatter_kernel(const int* __restrict__ adj,
                                                      const float* __restrict__ x,
                                                      float* __restrict__ agg,
                                                      float* __restrict__ cnt) {
    int e    = blockIdx.x * 8 + (threadIdx.x >> 5);
    int lane = threadIdx.x & 31;
    int src  = adj[e];
    int dst  = adj[NEDGE + e];
    const float* xs = x + (size_t)src * Hdim;
    float*       ag = agg + (size_t)dst * Hdim;
    #pragma unroll
    for (int j = 0; j < 8; ++j) {
        int d = lane + 32 * j;
        atomicAdd(&ag[d], xs[d]);
    }
    if (lane == 0) atomicAdd(&cnt[dst], 1.0f);
}

__global__ __launch_bounds__(256) void combine_kernel(const float* __restrict__ x,
                                                      const float* __restrict__ agg,
                                                      const float* __restrict__ cnt,
                                                      unsigned short* __restrict__ cat) {
    size_t idx = (size_t)blockIdx.x * 256 + threadIdx.x;
    int n = (int)(idx >> 8), d = (int)(idx & 255);
    cat[(size_t)n * 512 + d]       = f2bf(x[idx]);
    cat[(size_t)n * 512 + 256 + d] = f2bf(agg[idx] / fmaxf(cnt[n], 1.0f));
}

// ---------------------------------------------------------------------------
// Host launcher
// ---------------------------------------------------------------------------
extern "C" void kernel_launch(void* const* d_in, const int* in_sizes, int n_in,
                              void* d_out, int out_size, void* d_ws, size_t ws_size,
                              hipStream_t stream) {
    (void)in_sizes; (void)n_in; (void)out_size; (void)ws_size;

    const float* lit       = (const float*)d_in[0];
    const float* cf        = (const float*)d_in[1];
    const unsigned char* mask = (const unsigned char*)d_in[2];  // jax bool -> 1 byte
    const int*   adj       = (const int*)d_in[3];
    const float* pe_w1     = (const float*)d_in[4];
    const float* pe_b1     = (const float*)d_in[5];
    const float* pe_w2     = (const float*)d_in[6];
    const float* pe_b2     = (const float*)d_in[7];
    const float* attn_in_w = (const float*)d_in[8];
    const float* attn_in_b = (const float*)d_in[9];
    const float* attn_o_w  = (const float*)d_in[10];
    const float* attn_o_b  = (const float*)d_in[11];
    const float* attn_ln_g = (const float*)d_in[12];
    const float* attn_ln_b = (const float*)d_in[13];
    const float* gate_w    = (const float*)d_in[14];
    const float* gate_b    = (const float*)d_in[15];
    const float* trans_w   = (const float*)d_in[16];
    const float* trans_b   = (const float*)d_in[17];
    const float* fus_ln_g  = (const float*)d_in[18];
    const float* fus_ln_b  = (const float*)d_in[19];
    const float* mp_w      = (const float*)d_in[20];
    const float* mp_b      = (const float*)d_in[21];
    const float* mp_ln_g   = (const float*)d_in[22];
    const float* mp_ln_b   = (const float*)d_in[23];
    float* out = (float*)d_out;

    // Workspace arena
    char* w = (char*)d_ws;
    unsigned short* PW = (unsigned short*)w;     // packed bf16 weights
    unsigned short* pw2   = PW;                  //  65536
    unsigned short* pwq   = PW + 65536;
    unsigned short* pwk   = PW + 131072;
    unsigned short* pwv   = PW + 196608;
    unsigned short* pout  = PW + 262144;
    unsigned short* pgate = PW + 327680;         // 131072
    unsigned short* ptrans= PW + 458752;
    unsigned short* pmp0  = PW + 589824;
    unsigned short* pmp1  = PW + 720896;         // end 851968 elems

    const size_t SZF = (size_t)Ncl * Hdim * sizeof(float);   // 20,480,000 B
    char* base = w + (2u << 20);
    float* B0 = (float*)(base + 0 * SZF);
    float* B1 = (float*)(base + 1 * SZF);
    float* B2 = (float*)(base + 2 * SZF);
    float* B3 = (float*)(base + 3 * SZF);
    float* B4 = (float*)(base + 4 * SZF);
    float* B5 = (float*)(base + 5 * SZF);
    float* flags = (float*)(base + 6 * SZF);
    float* cntp  = flags + Ncl;

    // Buffer roles (reused across phases — liveness-checked)
    float* enc_f32  = B0;
    float* q_f32    = B1;
    unsigned short* ctx_bf = (unsigned short*)B2;
    float* litmean  = B3;
    float* ctxproj  = B4;
    float* attnrepr = B1;                         // q dead
    unsigned short* comb_bf = (unsigned short*)B5;
    float* gate_f   = B0;                         // enc dead
    float* trans_f  = B3;                         // litmean dead
    float* x_f32    = B4;                         // ctxproj dead
    float* agg_f    = B0;                         // gate dead
    unsigned short* cat_bf = (unsigned short*)B5; // comb dead
    float* upd_f    = B2;                         // ctx_bf dead
    unsigned short* hidden_bf = (unsigned short*)B5;                  // pre-comb
    unsigned short* enc_bf    = (unsigned short*)B5 + (size_t)Ncl * Hdim;

    const float qscale = 0.125f;   // 1/sqrt(DH=64)

    // 1) Pack all weight matrices to WMMA-fragment bf16 layout
    pack_kernel<<<256, 256, 0, stream>>>(pe_w2,     Hdim,     0,  8, pw2);
    pack_kernel<<<256, 256, 0, stream>>>(attn_in_w, 3 * Hdim, 0,  8, pwq);
    pack_kernel<<<256, 256, 0, stream>>>(attn_in_w, 3 * Hdim, Hdim,     8, pwk);
    pack_kernel<<<256, 256, 0, stream>>>(attn_in_w, 3 * Hdim, 2 * Hdim, 8, pwv);
    pack_kernel<<<256, 256, 0, stream>>>(attn_o_w,  Hdim,     0,  8, pout);
    pack_kernel<<<512, 256, 0, stream>>>(gate_w,    Hdim,     0, 16, pgate);
    pack_kernel<<<512, 256, 0, stream>>>(trans_w,   Hdim,     0, 16, ptrans);
    pack_kernel<<<512, 256, 0, stream>>>(mp_w,                Hdim, 0, 16, pmp0);
    pack_kernel<<<512, 256, 0, stream>>>(mp_w + 512 * Hdim,   Hdim, 0, 16, pmp1);

    // 2) Encoder: hidden = relu(cf@w1+b1); enc = hidden@w2+b2 (f32 + bf16)
    pe1_kernel<<<Ncl, 256, 0, stream>>>(cf, pe_w1, pe_b1, hidden_bf);
    gemm_wmma<8, 0><<<Ncl / 32, 256, 0, stream>>>(hidden_bf, pw2, pe_b2,
                                                  enc_f32, enc_bf, 1.0f);
    // 3) q = (enc@wq + bq) * 1/sqrt(DH)
    gemm_wmma<8, 1><<<Ncl / 32, 256, 0, stream>>>(enc_bf, pwq, attn_in_b,
                                                  q_f32, nullptr, qscale);
    // 4) Fused attention per clause (K/V via WMMA, masked softmax, ctx, lit_mean)
    attn_kernel<<<Ncl, 256, 0, stream>>>(lit, mask, q_f32, pwk, pwv, attn_in_b,
                                         ctx_bf, litmean, flags);
    // 5) ctx out-projection
    gemm_wmma<8, 0><<<Ncl / 32, 256, 0, stream>>>(ctx_bf, pout, attn_o_b,
                                                  ctxproj, nullptr, 1.0f);
    // 6) attn_repr = LN(masked ctxproj + enc) + lit_mean ; comb = [attn_repr, enc]
    attn_ln_kernel<<<Ncl / 8, 256, 0, stream>>>(ctxproj, enc_f32, litmean, flags,
                                                attn_ln_g, attn_ln_b,
                                                attnrepr, comb_bf);
    // 7) gate = sigmoid(comb@Wg+bg); transform = comb@Wt+bt
    gemm_wmma<16, 2><<<Ncl / 32, 256, 0, stream>>>(comb_bf, pgate, gate_b,
                                                   gate_f, nullptr, 1.0f);
    gemm_wmma<16, 0><<<Ncl / 32, 256, 0, stream>>>(comb_bf, ptrans, trans_b,
                                                   trans_f, nullptr, 1.0f);
    // 8) x = LN(gate*attn_repr + (1-gate)*transform)
    fuse_ln_kernel<<<Ncl / 8, 256, 0, stream>>>(attnrepr, gate_f, trans_f,
                                                fus_ln_g, fus_ln_b, x_f32);
    // 9) Two message-passing layers
    const size_t NH = (size_t)Ncl * Hdim;
    for (int i = 0; i < NLAY; ++i) {
        zero_kernel<<<(int)((NH + 255) / 256), 256, 0, stream>>>(agg_f, NH);
        zero_kernel<<<(Ncl + 255) / 256, 256, 0, stream>>>(cntp, (size_t)Ncl);
        scatter_kernel<<<NEDGE / 8, 256, 0, stream>>>(adj, x_f32, agg_f, cntp);
        combine_kernel<<<Ncl, 256, 0, stream>>>(x_f32, agg_f, cntp, cat_bf);
        gemm_wmma<16, 3><<<Ncl / 32, 256, 0, stream>>>(
            cat_bf, (i == 0 ? pmp0 : pmp1), mp_b + i * Hdim, upd_f, nullptr, 1.0f);
        mp_ln_kernel<<<Ncl / 8, 256, 0, stream>>>(
            x_f32, upd_f, mp_ln_g + i * Hdim, mp_ln_b + i * Hdim,
            (i == NLAY - 1) ? out : x_f32);
    }
}
